// MHSA_6519760355779
// MI455X (gfx1250) — compile-verified, matrix operand order checked by reading, not run on hardware
//
#include <hip/hip_runtime.h>

#define CC 256
#define NN 2304
#define BB 16
#define NB (NN / 64)   // 36 key blocks

typedef __attribute__((ext_vector_type(16))) __bf16        v16bf;
typedef __attribute__((ext_vector_type(8)))  float         v8f;
typedef __attribute__((ext_vector_type(4)))  unsigned int  v4u;

union Frag { v16bf f; v4u u[2]; };

__device__ __forceinline__ unsigned short f2bf(float x) {
    unsigned int u = __float_as_uint(x);
    u += 0x7FFFu + ((u >> 16) & 1u);          // round-to-nearest-even
    return (unsigned short)(u >> 16);
}

// Load one 16x32 bf16 A/B fragment slice for this lane.
// Layout (ISA 7.12.2): lane L holds M/N = L%16, K in [kh*8, kh*8+8) and
// [kh*8+16, kh*8+24) where kh = L/16.  p must point at (row, kh*8).
__device__ __forceinline__ Frag ldsFrag(const unsigned short* p) {
    Frag r;
    r.u[0] = *(const v4u*)(p);
    r.u[1] = *(const v4u*)(p + 16);
    return r;
}

__device__ __forceinline__ v8f bfwmma(const Frag& a, const Frag& b, v8f c) {
    return __builtin_amdgcn_wmma_f32_16x16x32_bf16(
        /*neg_a=*/false, a.f, /*neg_b=*/false, b.f,
        /*c_mod=*/(short)0, c, /*reuse_a=*/false, /*reuse_b=*/false);
}

// CDNA5 async memory->LDS copy, 16B per lane, GVS addressing, ASYNCcnt-tracked.
// Generic LDS pointers keep the LDS byte address in their low 32 bits (ISA 10.2).
__device__ __forceinline__ void asyncLoadB128(unsigned int ldsAddr,
                                              unsigned long long gbase,
                                              unsigned int voff) {
    asm volatile("global_load_async_to_lds_b128 %0, %1, %2"
                 :: "v"(ldsAddr), "v"(voff), "s"(gbase)
                 : "memory");
}
#define ASYNC_WAIT_16() asm volatile("s_wait_asynccnt 0x10" ::: "memory")
#define ASYNC_WAIT_0()  asm volatile("s_wait_asynccnt 0x0"  ::: "memory")

// ---------------------------------------------------------------------------
// Phase 1: projection  out[o,n] = sum_i W[o,i] * X[b,i,n] + bias[o]  (bf16 out)
// TRANS=1 -> out stored [b][n][o] (Q^T / K^T), else [b][o][n] (V)
// Full K=256 staged once -> 8 pipelined WMMAs, one barrier.
// ---------------------------------------------------------------------------
#define PRS 264   // padded LDS row stride (bf16): 132 dwords -> conflict-free

template <int TRANS>
__global__ __launch_bounds__(256)
void proj_kernel(const float* __restrict__ X, const float* __restrict__ W,
                 const float* __restrict__ bias, unsigned short* __restrict__ out) {
    __shared__ unsigned short Wl[32 * PRS];   // [o][c]
    __shared__ unsigned short Xl[64 * PRS];   // [n][c]

    const int b  = blockIdx.z;
    const int n0 = blockIdx.x * 64;
    const int o0 = blockIdx.y * 32;
    const int tid  = threadIdx.x;
    const int wave = tid >> 5, lane = tid & 31;
    const int laneQ = lane & 15, kh = lane >> 4;
    const int wo = wave >> 2, wn = wave & 3;   // 2 x 4 wave tile grid

    // stage W tile 32(o) x 256(c) as bf16
    {
        const int o = tid >> 3;
        #pragma unroll
        for (int p = 0; p < 8; ++p) {
            const int c = p * 32 + (tid & 7) * 4;
            float4 f = *(const float4*)(W + (size_t)(o0 + o) * CC + c);
            unsigned short* d = &Wl[o * PRS + c];
            d[0] = f2bf(f.x); d[1] = f2bf(f.y); d[2] = f2bf(f.z); d[3] = f2bf(f.w);
        }
    }
    // stage X tile transposed -> Xl[n][c]
    {
        const int nb = (tid & 7) * 8;
        #pragma unroll
        for (int p = 0; p < 8; ++p) {
            const int c = p * 32 + (tid >> 3);
            const float* g = X + ((size_t)b * CC + c) * NN + n0 + nb;
            float4 f0 = *(const float4*)g;
            float4 f1 = *(const float4*)(g + 4);
            Xl[(nb + 0) * PRS + c] = f2bf(f0.x);
            Xl[(nb + 1) * PRS + c] = f2bf(f0.y);
            Xl[(nb + 2) * PRS + c] = f2bf(f0.z);
            Xl[(nb + 3) * PRS + c] = f2bf(f0.w);
            Xl[(nb + 4) * PRS + c] = f2bf(f1.x);
            Xl[(nb + 5) * PRS + c] = f2bf(f1.y);
            Xl[(nb + 6) * PRS + c] = f2bf(f1.z);
            Xl[(nb + 7) * PRS + c] = f2bf(f1.w);
        }
    }
    __syncthreads();

    v8f acc = {};
    const unsigned short* wr = &Wl[(wo * 16 + laneQ) * PRS + kh * 8];
    const unsigned short* xr = &Xl[(wn * 16 + laneQ) * PRS + kh * 8];
    Frag a = ldsFrag(wr);
    Frag x = ldsFrag(xr);
    #pragma unroll
    for (int c0 = 0; c0 < CC; c0 += 32) {      // software-pipelined frag loads
        Frag an, xn;
        if (c0 + 32 < CC) {
            an = ldsFrag(wr + c0 + 32);
            xn = ldsFrag(xr + c0 + 32);
        }
        acc = bfwmma(a, x, acc);
        if (c0 + 32 < CC) { a = an; x = xn; }
    }

    const int n = n0 + wn * 16 + laneQ;
    #pragma unroll
    for (int r = 0; r < 8; ++r) {
        const int o = o0 + wo * 16 + kh * 8 + r;   // C/D layout: M = r + kh*8
        const unsigned short v = f2bf(acc[r] + bias[o]);
        if (TRANS) out[((size_t)b * NN + n) * CC + o] = v;
        else       out[((size_t)b * CC + o) * NN + n] = v;
    }
}

// ---------------------------------------------------------------------------
// Phase 2: flash attention, 64-query blocks, double-buffered async 64-key blocks.
// ---------------------------------------------------------------------------
#define SQK 264   // 256 + 8 pad (ushorts)
#define SVP 72    // 64 + 8 pad
#define SPP 72

#define SM_Q   0
#define SM_K0  (SM_Q  + 64 * SQK)
#define SM_K1  (SM_K0 + 64 * SQK)
#define SM_V0  (SM_K1 + 64 * SQK)
#define SM_V1  (SM_V0 + 256 * SVP)
#define SM_P   (SM_V1 + 256 * SVP)
#define SM_ST  (SM_P  + 64 * SPP)               // stats (floats), in ushort units
#define SMEM_BYTES (SM_ST * 2 + 512 * 4)        // 186368 bytes (<320KB WGP LDS)

// issue one 64-key block's K^T and V tiles as async-to-LDS (16 instrs/wave)
__device__ __forceinline__ void stageKV(unsigned long long kbase, unsigned long long vbase,
                                        unsigned short* sKb, unsigned short* sVb,
                                        int j0, int tid) {
    const int row = tid >> 2, part = tid & 3;
    #pragma unroll
    for (int i = 0; i < 8; ++i) {
        const int cseg = (part + i * 4) * 8;
        asyncLoadB128((unsigned int)(unsigned long long)&sKb[row * SQK + cseg],
                      kbase, (unsigned int)(((j0 + row) * CC + cseg) * 2));
    }
    #pragma unroll
    for (int i = 0; i < 8; ++i) {
        asyncLoadB128((unsigned int)(unsigned long long)&sVb[tid * SVP + i * 8],
                      vbase, (unsigned int)((tid * NN + j0 + i * 8) * 2));
    }
}

__global__ __launch_bounds__(256)
void attn_kernel(const unsigned short* __restrict__ Qt,
                 const unsigned short* __restrict__ Kt,
                 const unsigned short* __restrict__ Vm,
                 const float* __restrict__ X,
                 float* __restrict__ Out) {
    extern __shared__ unsigned short sm[];
    unsigned short* sQ  = sm + SM_Q;    // [64 q][SQK]  Q^T rows
    unsigned short* sK0 = sm + SM_K0;   // [64 j][SQK]  K^T rows (buf 0)
    unsigned short* sK1 = sm + SM_K1;
    unsigned short* sV0 = sm + SM_V0;   // [256 c][SVP] V rows (buf 0)
    unsigned short* sV1 = sm + SM_V1;
    unsigned short* sP  = sm + SM_P;    // [64 q][SPP]  probs bf16
    float* st   = (float*)(sm + SM_ST);
    float* rowM = st;          // running max   [64]
    float* rowL = st + 64;     // running sum   [64]
    float* rowA = st + 128;    // alpha         [64]
    float* rowN = st + 192;    // new max       [64]
    float* pMax = st + 256;    // partial max [2][64]
    float* pSum = st + 384;    // partial sum [2][64]

    const int b   = blockIdx.y;
    const int q0  = blockIdx.x * 64;
    const int tid = threadIdx.x;
    const int wave = tid >> 5, lane = tid & 31;
    const int laneQ = lane & 15, kh = lane >> 4;
    const int qtS = wave & 3, jt0 = (wave >> 2) * 2, half = wave >> 2;
    const int qtO = wave & 3, ctB = (wave >> 2) * 8;

    const unsigned long long qbase = (unsigned long long)(Qt + (size_t)b * NN * CC);
    const unsigned long long kbase = (unsigned long long)(Kt + (size_t)b * NN * CC);
    const unsigned long long vbase = (unsigned long long)(Vm + (size_t)b * CC * NN);

    if (tid < 64) { rowM[tid] = -3.0e38f; rowL[tid] = 0.0f; }

    // prologue: async-load Q block (8 instrs/wave) + key block 0 (16 instrs/wave)
    {
        const int row = tid >> 2, part = tid & 3;
        #pragma unroll
        for (int i = 0; i < 8; ++i) {
            const int cseg = (part + i * 4) * 8;
            asyncLoadB128((unsigned int)(unsigned long long)&sQ[row * SQK + cseg],
                          qbase, (unsigned int)(((q0 + row) * CC + cseg) * 2));
        }
    }
    stageKV(kbase, vbase, sK0, sV0, 0, tid);

    v8f oacc[8];
    { v8f z = {}; for (int i = 0; i < 8; ++i) oacc[i] = z; }

    for (int kb = 0; kb < NB; ++kb) {
        unsigned short* sKc = (kb & 1) ? sK1 : sK0;
        unsigned short* sVc = (kb & 1) ? sV1 : sV0;
        // prefetch next key block into the spare buffer, then retire current
        if (kb + 1 < NB) {
            stageKV(kbase, vbase, (kb & 1) ? sK0 : sK1, (kb & 1) ? sV0 : sV1,
                    (kb + 1) * 64, tid);
            ASYNC_WAIT_16();      // async loads complete in order
        } else {
            ASYNC_WAIT_0();
        }
        __syncthreads();          // every wave's current-block loads have landed

        // ---- S = Q^T K : each wave -> 16q x 32j (2 tiles), K-dim = 256 ----
        v8f s0v = {}, s1v = {};
        const unsigned short* qrow = sQ  + (qtS * 16 + laneQ) * SQK + kh * 8;
        const unsigned short* kr0  = sKc + ((jt0 + 0) * 16 + laneQ) * SQK + kh * 8;
        const unsigned short* kr1  = sKc + ((jt0 + 1) * 16 + laneQ) * SQK + kh * 8;
        Frag a  = ldsFrag(qrow);
        Frag b0 = ldsFrag(kr0);
        Frag b1 = ldsFrag(kr1);
        #pragma unroll
        for (int c0 = 0; c0 < CC; c0 += 32) {   // software-pipelined frag loads
            Frag an, b0n, b1n;
            if (c0 + 32 < CC) {
                an  = ldsFrag(qrow + c0 + 32);
                b0n = ldsFrag(kr0 + c0 + 32);
                b1n = ldsFrag(kr1 + c0 + 32);
            }
            s0v = bfwmma(a, b0, s0v);
            s1v = bfwmma(a, b1, s1v);
            if (c0 + 32 < CC) { a = an; b0 = b0n; b1 = b1n; }
        }

        // ---- partial row max (16 lanes hold one row; reduce within halves) ----
        #pragma unroll
        for (int r = 0; r < 8; ++r) {
            float mr = fmaxf(s0v[r], s1v[r]);
            mr = fmaxf(mr, __shfl_xor(mr, 1, 16));
            mr = fmaxf(mr, __shfl_xor(mr, 2, 16));
            mr = fmaxf(mr, __shfl_xor(mr, 4, 16));
            mr = fmaxf(mr, __shfl_xor(mr, 8, 16));
            if (laneQ == 0) pMax[half * 64 + qtS * 16 + kh * 8 + r] = mr;
        }
        __syncthreads();
        if (tid < 64) {
            const float bm = fmaxf(pMax[tid], pMax[64 + tid]);
            const float mo = rowM[tid];
            const float mn = fmaxf(mo, bm);
            rowN[tid] = mn;
            rowA[tid] = __expf(mo - mn);
            rowM[tid] = mn;
        }
        __syncthreads();

        // ---- P = exp(S - mn) -> LDS bf16, partial row sums ----
        #pragma unroll
        for (int r = 0; r < 8; ++r) {
            const int row = qtS * 16 + kh * 8 + r;
            const float mn = rowN[row];
            const float p0 = __expf(s0v[r] - mn);
            const float p1 = __expf(s1v[r] - mn);
            sP[row * SPP + (jt0 + 0) * 16 + laneQ] = f2bf(p0);
            sP[row * SPP + (jt0 + 1) * 16 + laneQ] = f2bf(p1);
            float s = p0 + p1;
            s += __shfl_xor(s, 1, 16);
            s += __shfl_xor(s, 2, 16);
            s += __shfl_xor(s, 4, 16);
            s += __shfl_xor(s, 8, 16);
            if (laneQ == 0) pSum[half * 64 + row] = s;
        }
        __syncthreads();
        if (tid < 64) rowL[tid] = rowL[tid] * rowA[tid] + pSum[tid] + pSum[64 + tid];

        // ---- O += V * P^T : wave -> 128c x 16q (8 tiles), per-lane rescale ----
        const int qloc = qtO * 16 + laneQ;
        const float sc = rowA[qloc];
        #pragma unroll
        for (int ci = 0; ci < 8; ++ci)
            #pragma unroll
            for (int r = 0; r < 8; ++r) oacc[ci][r] *= sc;

        {   // flattened 16-wmma chain, pipelined A-fragment loads
            const unsigned short* prow = sP + qloc * SPP + kh * 8;
            Frag bp0 = ldsFrag(prow);
            Frag bp1 = ldsFrag(prow + 32);
            Frag av = ldsFrag(sVc + (ctB * 16 + laneQ) * SVP + kh * 8);
            #pragma unroll
            for (int t = 0; t < 16; ++t) {
                const int jj = t >> 3, ci = t & 7;
                Frag an;
                if (t < 15) {
                    const int tn = t + 1;
                    an = ldsFrag(sVc + ((ctB + (tn & 7)) * 16 + laneQ) * SVP
                                 + (tn >> 3) * 32 + kh * 8);
                }
                oacc[ci] = bfwmma(av, jj ? bp1 : bp0, oacc[ci]);
                if (t < 15) av = an;
            }
        }
        __syncthreads();          // done reading this buffer / sP before next issue
    }

    // ---- normalize, add residual, store (coalesced along q) ----
    const int qloc = qtO * 16 + laneQ;
    const int gq = q0 + qloc;
    const float rl = 1.0f / rowL[qloc];
    #pragma unroll
    for (int ci = 0; ci < 8; ++ci) {
        #pragma unroll
        for (int r = 0; r < 8; ++r) {
            const int c = (ctB + ci) * 16 + kh * 8 + r;
            const size_t off = ((size_t)b * CC + c) * NN + gq;
            Out[off] = oacc[ci][r] * rl + X[off];
        }
    }
}

extern "C" void kernel_launch(void* const* d_in, const int* in_sizes, int n_in,
                              void* d_out, int out_size, void* d_ws, size_t ws_size,
                              hipStream_t stream) {
    const float* x  = (const float*)d_in[0];
    const float* wq = (const float*)d_in[1];
    const float* bq = (const float*)d_in[2];
    const float* wk = (const float*)d_in[3];
    const float* bk = (const float*)d_in[4];
    const float* wv = (const float*)d_in[5];
    const float* bv = (const float*)d_in[6];
    float* out = (float*)d_out;

    unsigned short* qt = (unsigned short*)d_ws;             // Q^T [b][n][c] bf16
    unsigned short* kt = qt + (size_t)BB * NN * CC;         // K^T [b][n][c] bf16
    unsigned short* vm = kt + (size_t)BB * NN * CC;         // V   [b][c][n] bf16

    (void)in_sizes; (void)n_in; (void)out_size; (void)ws_size;

    hipFuncSetAttribute((const void*)attn_kernel,
                        hipFuncAttributeMaxDynamicSharedMemorySize, SMEM_BYTES);

    dim3 pg(NN / 64, CC / 32, BB);
    proj_kernel<1><<<pg, 256, 0, stream>>>(x, wq, bq, qt);
    proj_kernel<1><<<pg, 256, 0, stream>>>(x, wk, bk, kt);
    proj_kernel<0><<<pg, 256, 0, stream>>>(x, wv, bv, vm);

    dim3 ag(NN / 64, BB);
    attn_kernel<<<ag, 256, SMEM_BYTES, stream>>>(qt, kt, vm, x, out);
}
